// SwinTransformerBlock_22170621182287
// MI455X (gfx1250) — matrix operand order repaired and checked by reading, not compiled, not run on hardware
//
#include <hip/hip_runtime.h>
#include <hip/hip_bf16.h>
#include <cmath>
#include <cstdint>

// ---------------------------------------------------------------------------
// Swin Transformer block (B=32, H=W=56, C=128, heads=4, ws=7, shift=3, mlp=4)
// WMMA f16 (16x16x32, f32 accum) for all GEMMs + window attention.
// TDM (tensor_load_to_lds) stages Q/K tiles for the non-shifted attention.
// ---------------------------------------------------------------------------

typedef __attribute__((ext_vector_type(16))) _Float16 v16h;
typedef __attribute__((ext_vector_type(8)))  _Float16 v8h;
typedef __attribute__((ext_vector_type(8)))  float    v8f;
typedef __attribute__((ext_vector_type(4)))  unsigned int u32x4;
typedef __attribute__((ext_vector_type(8)))  unsigned int u32x8;

#define NTOK   100352   // 32*56*56
#define CH     128
#define DH     32
#define HH     56
#define WW_    56
#define WS_    7
#define SS_    3

// ---- WMMA fragment loaders -------------------------------------------------
// A: 16x32 f16, row-major [M,K]. Lane (l&15)=row; l>>4 selects K-halves.
__device__ __forceinline__ v16h load_a_frag(const _Float16* A, int lda, int m0, int k0) {
    int lane = threadIdx.x & 31;
    const _Float16* p = A + (size_t)(m0 + (lane & 15)) * lda + k0 + ((lane >> 4) << 3);
    v8h lo = *(const v8h*)p;
    v8h hi = *(const v8h*)(p + 16);
    v16h r;
#pragma unroll
    for (int i = 0; i < 8; ++i) { r[i] = lo[i]; r[i + 8] = hi[i]; }
    return r;
}

// B: 32x16 f16 given as Bt row-major [N,K]. Lane (l&15)=col n; l>>4 selects K-halves.
__device__ __forceinline__ v16h load_b_frag(const _Float16* Bt, int ldb, int n0, int k0) {
    int lane = threadIdx.x & 31;
    const _Float16* p = Bt + (size_t)(n0 + (lane & 15)) * ldb + k0 + ((lane >> 4) << 3);
    v8h lo = *(const v8h*)p;
    v8h hi = *(const v8h*)(p + 16);
    v16h r;
#pragma unroll
    for (int i = 0; i < 8; ++i) { r[i] = lo[i]; r[i + 8] = hi[i]; }
    return r;
}

// ---- generic GEMM: C[M,N] = A[M,K](f16) * Bt[N,K](f16) + bias (+gelu)(+res)
// one wave computes a 64x64 tile; M,N multiples of 64; K multiple of 32.
// Epilogue configuration is compile-time to keep it branch-free.
template <bool GELU, bool RES, bool WF, bool WH>
__global__ __launch_bounds__(32) void gemm_wmma(
    const _Float16* __restrict__ A, const _Float16* __restrict__ Bt,
    const float* __restrict__ bias, const float* __restrict__ resid,
    float* __restrict__ outF, _Float16* __restrict__ outH,
    int M, int N, int K)
{
    int tiles_n = N >> 6;
    int tn = blockIdx.x % tiles_n;
    int tm = blockIdx.x / tiles_n;
    int m0 = tm << 6, n0 = tn << 6;
    int lane = threadIdx.x & 31;

    // pull the weight panel toward the caches early (global_prefetch_b8)
    __builtin_prefetch(Bt + (size_t)(n0 + lane) * K, 0, 1);

    v8f zero = {};
    v8f acc[4][4];
#pragma unroll
    for (int i = 0; i < 4; ++i)
#pragma unroll
        for (int j = 0; j < 4; ++j) acc[i][j] = zero;

    for (int k0 = 0; k0 < K; k0 += 32) {
        v16h af[4], bf[4];
#pragma unroll
        for (int i = 0; i < 4; ++i) af[i] = load_a_frag(A, K, m0 + i * 16, k0);
#pragma unroll
        for (int j = 0; j < 4; ++j) bf[j] = load_b_frag(Bt, K, n0 + j * 16, k0);
#pragma unroll
        for (int i = 0; i < 4; ++i)
#pragma unroll
            for (int j = 0; j < 4; ++j)
                acc[i][j] = __builtin_amdgcn_wmma_f32_16x16x32_f16(
                    false, af[i], false, bf[j], (short)0, acc[i][j], false, false);
    }

    int nn = lane & 15, g = lane >> 4;
#pragma unroll
    for (int i = 0; i < 4; ++i) {
#pragma unroll
        for (int j = 0; j < 4; ++j) {
            int n = n0 + j * 16 + nn;
            float bn = bias[n];
#pragma unroll
            for (int r = 0; r < 8; ++r) {
                int m = m0 + i * 16 + g * 8 + r;
                float v = acc[i][j][r] + bn;
                if (GELU) v = 0.5f * v * (1.f + erff(v * 0.70710678118654752f));
                size_t idx = (size_t)m * N + n;
                if (RES) v += resid[idx];
                if (WF) outF[idx] = v;
                if (WH) outH[idx] = (_Float16)v;
            }
        }
    }
}

// ---- LayerNorm: 1 wave per token (128 ch), f32 in, f16 out -----------------
__global__ __launch_bounds__(128) void ln_f16(
    const float* __restrict__ x, const float* __restrict__ gw,
    const float* __restrict__ bw, _Float16* __restrict__ out)
{
    int t = blockIdx.x * 4 + (threadIdx.x >> 5);
    int lane = threadIdx.x & 31;
    float4 v = ((const float4*)(x + (size_t)t * CH))[lane];
    float s = v.x + v.y + v.z + v.w;
#pragma unroll
    for (int m = 1; m < 32; m <<= 1) s += __shfl_xor(s, m, 32);
    float mu = s * (1.f / 128.f);
    float dx = v.x - mu, dy = v.y - mu, dz = v.z - mu, dw = v.w - mu;
    float q = dx * dx + dy * dy + dz * dz + dw * dw;
#pragma unroll
    for (int m = 1; m < 32; m <<= 1) q += __shfl_xor(q, m, 32);
    float rs = rsqrtf(q * (1.f / 128.f) + 1e-5f);
    int c = lane << 2;
    float4 gg = ((const float4*)gw)[lane];
    float4 bb = ((const float4*)bw)[lane];
    _Float16* o = out + (size_t)t * CH + c;
    o[0] = (_Float16)(dx * rs * gg.x + bb.x);
    o[1] = (_Float16)(dy * rs * gg.y + bb.y);
    o[2] = (_Float16)(dz * rs * gg.z + bb.z);
    o[3] = (_Float16)(dw * rs * gg.w + bb.w);
}

// ---- weight convert: fp32 W[K,N] -> f16 Wt[N,K] ----------------------------
__global__ void wconv(const float* __restrict__ W, _Float16* __restrict__ Wt,
                      int K, int N)
{
    int idx = blockIdx.x * 256 + threadIdx.x;
    if (idx >= N * K) return;
    int n = idx / K, k = idx - n * K;
    Wt[idx] = (_Float16)W[(size_t)k * N + n];
}

// ---- TDM: load a [7 rows x 7 tokens x 32 halfs] tile into LDS --------------
// D# per CDNA5 ISA ch.8: group0 = {count/lds/global/type}, group1 = dims/tile,
// groups 2/3 present (3-D tile). data_size=2B, dim0_stride=384 halfs (token),
// dim1_stride=56*384 halfs (image row). 49 rows * 64B land contiguously.
__device__ __forceinline__ void tdm_load_win_tile(uint32_t lds_off, uint64_t gaddr) {
    u32x4 g0;
    g0[0] = 1u;                                    // count=1, user descriptor
    g0[1] = lds_off;                               // LDS byte address
    g0[2] = (uint32_t)gaddr;                       // global_addr[31:0]
    g0[3] = ((uint32_t)(gaddr >> 32) & 0x1FFFFFFu) | (2u << 30);  // [56:32]|type=2
    u32x8 g1;
    g1[0] = 1u << 16;                              // data_size = 1 (2 bytes)
    g1[1] = 32u << 16;                             // tensor_dim0 = 32
    g1[2] = 0x8000u << 16;                         // tensor_dim1 = 32768
    g1[3] = 32u << 16;                             // tile_dim0 = 32
    g1[4] = 7u | (7u << 16);                       // tile_dim1 = 7, tile_dim2 = 7
    g1[5] = 384u;                                  // tensor_dim0_stride = 384
    g1[6] = 21504u << 16;                          // tensor_dim1_stride = 56*384
    g1[7] = 0u;
    u32x4 g2;
    g2[0] = 0x8000u;                               // tensor_dim2 = 32768
    g2[1] = 0u; g2[2] = 0u; g2[3] = 0u;            // tile_dim3 = 0 (unused)
    u32x4 g3 = {0u, 0u, 0u, 0u};
    asm volatile("tensor_load_to_lds %0, %1, %2, %3"
                 :: "s"(g0), "s"(g1), "s"(g2), "s"(g3)
                 : "memory");
}

// ---- windowed attention: 1 wave per (window, head) -------------------------
// qkv: f16 [token, 384]  (Q at c, K at 128+c, V at 256+c, c = head*32+d)
// out: f16 [token, 128]
template <int SHIFTED>
__global__ __launch_bounds__(32) void attn_wmma(
    const _Float16* __restrict__ qkv, _Float16* __restrict__ out)
{
    __shared__ _Float16 sQ[64 * 32];
    __shared__ _Float16 sK[64 * 32];
    __shared__ _Float16 sVt[32 * 64];
    __shared__ _Float16 sS[64 * 64];
    __shared__ int sId[64];

    int bid = blockIdx.x;
    int head = bid & 3;
    int win = bid >> 2;
    int b = win >> 6;
    int w64 = win & 63;
    int wh = w64 >> 3, wwi = w64 & 7;
    int lane = threadIdx.x;

    v8h hz = {};
    if (!SHIFTED) {
        // ---- TDM path: Q and K tiles are clean strided 3-D tiles ----------
        size_t origin = ((size_t)b * 3136 + (size_t)(wh * 7) * WW_ + (wwi * 7)) * 384
                      + (size_t)head * DH;
        tdm_load_win_tile((uint32_t)(uintptr_t)&sQ[0],
                          (uint64_t)(uintptr_t)(qkv + origin));
        tdm_load_win_tile((uint32_t)(uintptr_t)&sK[0],
                          (uint64_t)(uintptr_t)(qkv + origin + 128));
        // V gathered manually (needs transpose for the B-fragment layout)
        for (int r = lane; r < 64; r += 32) {
            if (r < 49) {
                int rr = r / 7, cc = r - rr * 7;
                size_t t = (size_t)b * 3136 + (size_t)(wh * 7 + rr) * WW_ + (wwi * 7 + cc);
                const _Float16* pv = qkv + t * 384 + head * DH + 256;
#pragma unroll
                for (int d = 0; d < 32; ++d) sVt[d * 64 + r] = pv[d];
            } else {
#pragma unroll
                for (int d8 = 0; d8 < 4; ++d8) {
                    *(v8h*)&sQ[r * 32 + d8 * 8] = hz;   // zero pad rows 49..63
                    *(v8h*)&sK[r * 32 + d8 * 8] = hz;
                }
#pragma unroll
                for (int d = 0; d < 32; ++d) sVt[d * 64 + r] = (_Float16)0.f;
            }
        }
        __builtin_amdgcn_s_wait_tensorcnt(0);
    } else {
        // ---- shifted: window wraps mod 56, gather manually ----------------
        for (int r = lane; r < 64; r += 32) {
            int id = 9;
            if (r < 49) {
                int rr = r / 7, cc = r - rr * 7;
                int hr = wh * 7 + rr, wr = wwi * 7 + cc;
                int h = (hr + HH - SS_) % HH;          // undo roll(+3)
                int w = (wr + WW_ - SS_) % WW_;
                int rh = (hr < HH - WS_) ? 0 : ((hr < HH - SS_) ? 1 : 2);
                int rw = (wr < WW_ - WS_) ? 0 : ((wr < WW_ - SS_) ? 1 : 2);
                id = rh * 3 + rw;
                size_t base = ((size_t)b * 3136 + (size_t)h * WW_ + w) * 384 + head * DH;
                const _Float16* pq = qkv + base;
#pragma unroll
                for (int d8 = 0; d8 < 4; ++d8) {
                    *(v8h*)&sQ[r * 32 + d8 * 8] = *(const v8h*)(pq + d8 * 8);
                    *(v8h*)&sK[r * 32 + d8 * 8] = *(const v8h*)(pq + 128 + d8 * 8);
                }
#pragma unroll
                for (int d = 0; d < 32; ++d) sVt[d * 64 + r] = pq[256 + d];
            } else {
#pragma unroll
                for (int d8 = 0; d8 < 4; ++d8) {
                    *(v8h*)&sQ[r * 32 + d8 * 8] = hz;
                    *(v8h*)&sK[r * 32 + d8 * 8] = hz;
                }
#pragma unroll
                for (int d = 0; d < 32; ++d) sVt[d * 64 + r] = (_Float16)0.f;
            }
            sId[r] = id;
        }
    }
    __syncthreads();  // single wave: lowers to s_nop; keeps LDS ordering clear

    const float scale = 0.17677669529663687f; // 1/sqrt(32)
    int nn = lane & 15, g = lane >> 4;
    v8f zero = {};

    // S = softmax(scale*Q*K^T + mask) -> sS (f16)
#pragma unroll
    for (int mt = 0; mt < 4; ++mt) {
        v16h a = load_a_frag(sQ, 32, mt * 16, 0);
        v8f c[4];
#pragma unroll
        for (int nt = 0; nt < 4; ++nt) {
            v16h bf = load_b_frag(sK, 32, nt * 16, 0);
            c[nt] = __builtin_amdgcn_wmma_f32_16x16x32_f16(
                false, a, false, bf, (short)0, zero, false, false);
        }
#pragma unroll
        for (int j = 0; j < 8; ++j) {
            int m = mt * 16 + g * 8 + j;
            int idm = SHIFTED ? sId[m] : 0;
            float best = -1e30f;
#pragma unroll
            for (int nt = 0; nt < 4; ++nt) {
                int n = nt * 16 + nn;
                float v = c[nt][j] * scale;
                if (n >= 49) v = -1e30f;
                else if (SHIFTED && sId[n] != idm) v += -100.f;
                c[nt][j] = v;
                best = fmaxf(best, v);
            }
#pragma unroll
            for (int msk = 1; msk < 16; msk <<= 1)
                best = fmaxf(best, __shfl_xor(best, msk, 32));
            float ssum = 0.f;
#pragma unroll
            for (int nt = 0; nt < 4; ++nt) {
                float e = expf(c[nt][j] - best);
                c[nt][j] = e;
                ssum += e;
            }
#pragma unroll
            for (int msk = 1; msk < 16; msk <<= 1)
                ssum += __shfl_xor(ssum, msk, 32);
            float inv = 1.f / ssum;
#pragma unroll
            for (int nt = 0; nt < 4; ++nt)
                sS[m * 64 + nt * 16 + nn] = (_Float16)(c[nt][j] * inv);
        }
    }
    __syncthreads();

    // O = S * V  (64x64 @ 64x32), scatter back to token order
#pragma unroll
    for (int mt = 0; mt < 4; ++mt) {
        v8f o[2] = {zero, zero};
#pragma unroll
        for (int ks = 0; ks < 2; ++ks) {
            v16h a = load_a_frag(sS, 64, mt * 16, ks * 32);
#pragma unroll
            for (int nt = 0; nt < 2; ++nt) {
                v16h bf = load_b_frag(sVt, 64, nt * 16, ks * 32);
                o[nt] = __builtin_amdgcn_wmma_f32_16x16x32_f16(
                    false, a, false, bf, (short)0, o[nt], false, false);
            }
        }
#pragma unroll
        for (int nt = 0; nt < 2; ++nt) {
            int d = nt * 16 + nn;
#pragma unroll
            for (int j = 0; j < 8; ++j) {
                int m = mt * 16 + g * 8 + j;
                if (m < 49) {
                    int rr = m / 7, cc = m - rr * 7;
                    int hr = wh * 7 + rr, wr = wwi * 7 + cc;
                    int h = hr, w = wr;
                    if (SHIFTED) { h = (hr + HH - SS_) % HH; w = (wr + WW_ - SS_) % WW_; }
                    size_t t = (size_t)b * 3136 + (size_t)h * WW_ + w;
                    out[t * CH + head * DH + d] = (_Float16)o[nt][j];
                }
            }
        }
    }
}

// ---------------------------------------------------------------------------
extern "C" void kernel_launch(void* const* d_in, const int* in_sizes, int n_in,
                              void* d_out, int out_size, void* d_ws, size_t ws_size,
                              hipStream_t stream) {
    (void)in_sizes; (void)n_in; (void)out_size; (void)ws_size;
    const float* x      = (const float*)d_in[0];
    const float* g1     = (const float*)d_in[1];
    const float* be1    = (const float*)d_in[2];
    const float* g2     = (const float*)d_in[3];
    const float* be2    = (const float*)d_in[4];
    const float* g3     = (const float*)d_in[5];
    const float* be3    = (const float*)d_in[6];
    const float* g4     = (const float*)d_in[7];
    const float* be4    = (const float*)d_in[8];
    const float* qkv_w1 = (const float*)d_in[9];
    const float* qkv_b1 = (const float*)d_in[10];
    const float* proj_w1= (const float*)d_in[11];
    const float* proj_b1= (const float*)d_in[12];
    const float* qkv_w2 = (const float*)d_in[13];
    const float* qkv_b2 = (const float*)d_in[14];
    const float* proj_w2= (const float*)d_in[15];
    const float* proj_b2= (const float*)d_in[16];
    const float* fw1    = (const float*)d_in[17];
    const float* fb1    = (const float*)d_in[18];
    const float* fw2    = (const float*)d_in[19];
    const float* fb2    = (const float*)d_in[20];
    const float* fw3    = (const float*)d_in[21];
    const float* fb3    = (const float*)d_in[22];
    const float* fw4    = (const float*)d_in[23];
    const float* fb4    = (const float*)d_in[24];
    float* out = (float*)d_out;

    _Float16* wsH = (_Float16*)d_ws;
    _Float16* wqkv1t  = wsH;                 // 384*128
    _Float16* wproj1t = wqkv1t  + 49152;     // 128*128
    _Float16* wf1t    = wproj1t + 16384;     // 512*128
    _Float16* wf2t    = wf1t    + 65536;     // 128*512
    _Float16* wqkv2t  = wf2t    + 65536;
    _Float16* wproj2t = wqkv2t  + 49152;
    _Float16* wf3t    = wproj2t + 16384;
    _Float16* wf4t    = wf3t    + 65536;
    _Float16* buf0    = wf4t    + 65536;              // [NTOK,128] f16
    _Float16* buf1    = buf0 + (size_t)NTOK * 128;    // [NTOK,384] f16
    _Float16* buf2    = buf1 + (size_t)NTOK * 384;    // [NTOK,512] f16

    // weight transpose+convert (fp32 [K,N] -> f16 [N,K])
    wconv<<<(384*128+255)/256, 256, 0, stream>>>(qkv_w1,  wqkv1t,  128, 384);
    wconv<<<(128*128+255)/256, 256, 0, stream>>>(proj_w1, wproj1t, 128, 128);
    wconv<<<(128*512+255)/256, 256, 0, stream>>>(fw1,     wf1t,    128, 512);
    wconv<<<(512*128+255)/256, 256, 0, stream>>>(fw2,     wf2t,    512, 128);
    wconv<<<(384*128+255)/256, 256, 0, stream>>>(qkv_w2,  wqkv2t,  128, 384);
    wconv<<<(128*128+255)/256, 256, 0, stream>>>(proj_w2, wproj2t, 128, 128);
    wconv<<<(128*512+255)/256, 256, 0, stream>>>(fw3,     wf3t,    128, 512);
    wconv<<<(512*128+255)/256, 256, 0, stream>>>(fw4,     wf4t,    512, 128);

    const int LNG   = NTOK / 4;          // LN grid (4 tokens/block)
    const int TQKV  = (NTOK/64) * (384/64);
    const int TPROJ = (NTOK/64) * (128/64);
    const int TFFN1 = (NTOK/64) * (512/64);
    const int NATTN = 32 * 64 * 4;       // B * windows * heads

    // ---- sub-block 1: W-MSA ----
    ln_f16<<<LNG, 128, 0, stream>>>(x, g1, be1, buf0);
    gemm_wmma<false,false,false,true><<<TQKV, 32, 0, stream>>>(
        buf0, wqkv1t, qkv_b1, nullptr, nullptr, buf1, NTOK, 384, 128);
    attn_wmma<0><<<NATTN, 32, 0, stream>>>(buf1, buf2);
    gemm_wmma<false,true,true,false><<<TPROJ, 32, 0, stream>>>(
        buf2, wproj1t, proj_b1, x, out, nullptr, NTOK, 128, 128);
    // FFN 1
    ln_f16<<<LNG, 128, 0, stream>>>(out, g2, be2, buf0);
    gemm_wmma<true,false,false,true><<<TFFN1, 32, 0, stream>>>(
        buf0, wf1t, fb1, nullptr, nullptr, buf2, NTOK, 512, 128);
    gemm_wmma<false,true,true,false><<<TPROJ, 32, 0, stream>>>(
        buf2, wf2t, fb2, out, out, nullptr, NTOK, 128, 512);
    // ---- sub-block 2: SW-MSA ----
    ln_f16<<<LNG, 128, 0, stream>>>(out, g3, be3, buf0);
    gemm_wmma<false,false,false,true><<<TQKV, 32, 0, stream>>>(
        buf0, wqkv2t, qkv_b2, nullptr, nullptr, buf1, NTOK, 384, 128);
    attn_wmma<1><<<NATTN, 32, 0, stream>>>(buf1, buf2);
    gemm_wmma<false,true,true,false><<<TPROJ, 32, 0, stream>>>(
        buf2, wproj2t, proj_b2, out, out, nullptr, NTOK, 128, 128);
    // FFN 2
    ln_f16<<<LNG, 128, 0, stream>>>(out, g4, be4, buf0);
    gemm_wmma<true,false,false,true><<<TFFN1, 32, 0, stream>>>(
        buf0, wf3t, fb3, nullptr, nullptr, buf2, NTOK, 512, 128);
    gemm_wmma<false,true,true,false><<<TPROJ, 32, 0, stream>>>(
        buf2, wf4t, fb4, out, out, nullptr, NTOK, 128, 512);
}
